// Transformer_82540681494680
// MI455X (gfx1250) — matrix operand order repaired and checked by reading, not compiled, not run on hardware
//
#include <hip/hip_runtime.h>
#include <math.h>

// ---------------------------------------------------------------------------
// Types for CDNA5 WMMA (wave32): V_WMMA_F32_16X16X32_BF16
// ---------------------------------------------------------------------------
typedef __attribute__((ext_vector_type(16))) __bf16 bf16x16;
typedef __attribute__((ext_vector_type(8)))  __bf16 bf16x8;
typedef __attribute__((ext_vector_type(8)))  float  f32x8;

#define WMMA_BF16(A, B, C) \
  __builtin_amdgcn_wmma_f32_16x16x32_bf16(false, (A), false, (B), (short)0, (C), false, false)

// ---------------------------------------------------------------------------
// CDNA5 async global->LDS copy (ASYNCcnt-tracked), with safe fallback.
// Builtin signature (from compiler diagnostic): params are pointers to
// GCC-style int 4-vectors in AS(1)/AS(3).
// ---------------------------------------------------------------------------
#if defined(__has_builtin)
#if __has_builtin(__builtin_amdgcn_global_load_async_to_lds_b128) && \
    __has_builtin(__builtin_amdgcn_s_wait_asynccnt)
#define HAVE_ASYNC_LDS 1
#endif
#endif
#ifndef HAVE_ASYNC_LDS
#define HAVE_ASYNC_LDS 0
#endif

#if HAVE_ASYNC_LDS
typedef int v4i_g __attribute__((vector_size(16)));
typedef __attribute__((address_space(1))) v4i_g* gbl_v4p;
typedef __attribute__((address_space(3))) v4i_g* lds_v4p;
__device__ __forceinline__ lds_v4p to_lds(const void* p) {
  // generic LDS pointer: low 32 bits are the LDS byte offset
  return (lds_v4p)(unsigned int)(unsigned long long)p;
}
__device__ __forceinline__ gbl_v4p to_gbl(const void* p) {
  return (gbl_v4p)(unsigned long long)p;
}
#define ASYNC_CP16(l, g) \
  __builtin_amdgcn_global_load_async_to_lds_b128(to_gbl(g), to_lds(l), 0, 0)
#define WAIT_ASYNC(n) __builtin_amdgcn_s_wait_asynccnt(n)
#else
#define ASYNC_CP16(l, g) (*(bf16x8*)(l) = *(const bf16x8*)(g))
#define WAIT_ASYNC(n)
#endif

__device__ __forceinline__ bf16x16 frag_ld(const __bf16* p0, const __bf16* p1) {
  bf16x8 a = *(const bf16x8*)p0;   // 16-byte ds/global load
  bf16x8 b = *(const bf16x8*)p1;
  bf16x16 r;
#pragma unroll
  for (int i = 0; i < 8; ++i) { r[i] = a[i]; r[i + 8] = b[i]; }
  return r;
}

__device__ __forceinline__ f32x8 zero8() {
  f32x8 z;
#pragma unroll
  for (int i = 0; i < 8; ++i) z[i] = 0.f;
  return z;
}

// 16-lane-half reductions (C-matrix rows live in 16-lane halves; xor masks <16
// never cross the half boundary on wave32).
__device__ __forceinline__ float red_max16(float v) {
#pragma unroll
  for (int m = 1; m < 16; m <<= 1) v = fmaxf(v, __shfl_xor(v, m, 32));
  return v;
}
__device__ __forceinline__ float red_sum16(float v) {
#pragma unroll
  for (int m = 1; m < 16; m <<= 1) v += __shfl_xor(v, m, 32);
  return v;
}

__device__ __forceinline__ float block_sum256(float v, float* sred) {
#pragma unroll
  for (int m = 1; m < 32; m <<= 1) v += __shfl_xor(v, m, 32);
  __syncthreads();                       // protect prior readers of sred
  if ((threadIdx.x & 31) == 0) sred[threadIdx.x >> 5] = v;
  __syncthreads();
  float t = 0.f;
#pragma unroll
  for (int i = 0; i < 8; ++i) t += sred[i];
  return t;
}

// ---------------------------------------------------------------------------
// Weight convert + transpose: src fp32 [K][N] -> dst bf16 [N][K]  (W^T)
// ---------------------------------------------------------------------------
__global__ __launch_bounds__(256) void wconv_tr(const float* __restrict__ src,
                                                __bf16* __restrict__ dst,
                                                int K, int N) {
  __shared__ float tile[32][33];
  const int k0 = blockIdx.y * 32, n0 = blockIdx.x * 32;
  const int tr = threadIdx.x >> 5, tc = threadIdx.x & 31;
#pragma unroll
  for (int j = 0; j < 4; ++j) {
    int kk = tr + j * 8;
    tile[kk][tc] = src[(size_t)(k0 + kk) * N + n0 + tc];
  }
  __syncthreads();
#pragma unroll
  for (int j = 0; j < 4; ++j) {
    int nn = tr + j * 8;
    dst[(size_t)(n0 + nn) * K + k0 + tc] = (__bf16)tile[tc][nn];
  }
}

// ---------------------------------------------------------------------------
// LayerNorm (reference: var = sum((x-m)^2)/(D-1); y = g*(x-m)/(sqrt(var)+eps)+b)
// fp32 in -> bf16 out.  One row per block.
// ---------------------------------------------------------------------------
__global__ __launch_bounds__(256) void ln_to_bf16(const float* __restrict__ X,
                                                  const float* __restrict__ g,
                                                  const float* __restrict__ bta,
                                                  __bf16* __restrict__ Y, int D) {
  __shared__ float sred[8];
  const int row = blockIdx.x, tid = threadIdx.x;
  const float* x = X + (size_t)row * D;
  float ps = 0.f;
  for (int d = tid; d < D; d += 256) ps += x[d];
  const float mean = block_sum256(ps, sred) / (float)D;
  float pv = 0.f;
  for (int d = tid; d < D; d += 256) { float t = x[d] - mean; pv += t * t; }
  const float var = block_sum256(pv, sred) / (float)(D - 1);
  const float rstd = 1.f / (sqrtf(var) + 1e-6f);
  __bf16* y = Y + (size_t)row * D;
  for (int d = tid; d < D; d += 256)
    y[d] = (__bf16)(g[d] * (x[d] - mean) * rstd + bta[d]);
}

// ---------------------------------------------------------------------------
// GEMM: Out = act(A[MxK](bf16) * Wt[NxK]^T(bf16) + bias[N])
// mode 0: bf16 out;  mode 1: bf16 ReLU out;  mode 2: f32 out = Resid + v (in place)
// Block tile 128x128x64, 256 threads = 8 waves (4 in M x 2 in N), wave 32x64.
// Double-buffered LDS; next tile staged with GLOBAL_LOAD_ASYNC_TO_LDS_B128.
// ---------------------------------------------------------------------------
#define BM 128
#define BN 128
#define BK 64
#define LDT 72   // BK + 8 pad; 144-byte stride: 16B aligned, spreads LDS banks

__global__ __launch_bounds__(256) void gemm_bf16_wmma(
    const __bf16* __restrict__ A, const __bf16* __restrict__ Wt,
    const float* __restrict__ bias, void* __restrict__ Out,
    float* __restrict__ Resid, int M, int N, int K, int mode) {
  __shared__ __bf16 sA[2][BM][LDT];
  __shared__ __bf16 sB[2][BN][LDT];
  const int tid = threadIdx.x;
  const int lane = tid & 31, w = tid >> 5;
  const int ll = lane & 15, hi = lane >> 4;
  const int m0 = blockIdx.y * BM, n0 = blockIdx.x * BN;
  const int wm = (w & 3) * 32, wn = (w >> 2) * 64;

  f32x8 acc[2][4];
#pragma unroll
  for (int im = 0; im < 2; ++im)
#pragma unroll
    for (int in = 0; in < 4; ++in) acc[im][in] = zero8();

  // staging: each thread owns 64B of the A tile and 64B of the B tile per stage
  const int rs = tid >> 1, cs = (tid & 1) * 32;
  const size_t arow = (size_t)(m0 + rs) * K;
  const size_t brow = (size_t)(n0 + rs) * K;

#define STAGE(buf, k0)                                                        \
  {                                                                           \
    const __bf16* ga = &A[arow + (size_t)(k0) + cs];                          \
    const __bf16* gb = &Wt[brow + (size_t)(k0) + cs];                         \
    __bf16* la = &sA[buf][rs][cs];                                            \
    __bf16* lb = &sB[buf][rs][cs];                                            \
    ASYNC_CP16(la + 0, ga + 0);   ASYNC_CP16(la + 8, ga + 8);                 \
    ASYNC_CP16(la + 16, ga + 16); ASYNC_CP16(la + 24, ga + 24);               \
    ASYNC_CP16(lb + 0, gb + 0);   ASYNC_CP16(lb + 8, gb + 8);                 \
    ASYNC_CP16(lb + 16, gb + 16); ASYNC_CP16(lb + 24, gb + 24);               \
  }

  const int nk = K / BK;
  STAGE(0, 0);
  for (int kt = 0; kt < nk; ++kt) {
    const int cur = kt & 1;
    if (kt + 1 < nk) {
      STAGE(cur ^ 1, (kt + 1) * BK);
      WAIT_ASYNC(8);   // 8 in flight for next stage; previous stage landed
    } else {
      WAIT_ASYNC(0);
    }
    __syncthreads();

#pragma unroll
    for (int c = 0; c < 2; ++c) {          // two K=32 chunks inside the tile
      bf16x16 af[2], bfg[4];
#pragma unroll
      for (int im = 0; im < 2; ++im) {
        const __bf16* p = &sA[cur][wm + im * 16 + ll][c * 32];
        af[im] = frag_ld(p + hi * 8, p + 16 + hi * 8);     // A 16x32 layout
      }
#pragma unroll
      for (int in = 0; in < 4; ++in) {
        const __bf16* p = &sB[cur][wn + in * 16 + ll][c * 32];
        bfg[in] = frag_ld(p + hi * 8, p + 16 + hi * 8);    // B 32x16 layout
      }
#pragma unroll
      for (int im = 0; im < 2; ++im)
#pragma unroll
        for (int in = 0; in < 4; ++in)
          acc[im][in] = WMMA_BF16(af[im], bfg[in], acc[im][in]);
    }
    __syncthreads();
  }
#undef STAGE

  // Epilogue. C layout: VGPR i, lanes 0-15 -> row i; lanes 16-31 -> row i+8; col = ll.
  float bn[4];
#pragma unroll
  for (int in = 0; in < 4; ++in) bn[in] = bias[n0 + wn + in * 16 + ll];

  if (mode == 2) {
    float* o = (float*)Out;
#pragma unroll
    for (int im = 0; im < 2; ++im)
#pragma unroll
      for (int i = 0; i < 8; ++i) {
        const size_t mrow = (size_t)(m0 + wm + im * 16 + i + 8 * hi) * N;
#pragma unroll
        for (int in = 0; in < 4; ++in) {
          const size_t idx = mrow + n0 + wn + in * 16 + ll;
          o[idx] = Resid[idx] + acc[im][in][i] + bn[in];
        }
      }
  } else if (mode == 1) {
    __bf16* o = (__bf16*)Out;
#pragma unroll
    for (int im = 0; im < 2; ++im)
#pragma unroll
      for (int i = 0; i < 8; ++i) {
        const size_t mrow = (size_t)(m0 + wm + im * 16 + i + 8 * hi) * N;
#pragma unroll
        for (int in = 0; in < 4; ++in)
          o[mrow + n0 + wn + in * 16 + ll] =
              (__bf16)fmaxf(acc[im][in][i] + bn[in], 0.f);
      }
  } else {
    __bf16* o = (__bf16*)Out;
#pragma unroll
    for (int im = 0; im < 2; ++im)
#pragma unroll
      for (int i = 0; i < 8; ++i) {
        const size_t mrow = (size_t)(m0 + wm + im * 16 + i + 8 * hi) * N;
#pragma unroll
        for (int in = 0; in < 4; ++in)
          o[mrow + n0 + wn + in * 16 + ll] = (__bf16)(acc[im][in][i] + bn[in]);
      }
  }
}

// ---------------------------------------------------------------------------
// Flash attention: Q,K,V bf16 in [B,S,H*dk]; out bf16 same layout. dk=64.
// Block = 4 waves = 64 query rows for one (b,h). Online softmax over 32-key chunks.
// ---------------------------------------------------------------------------
__global__ __launch_bounds__(128) void attn_wmma(
    const __bf16* __restrict__ Q, const __bf16* __restrict__ Kb,
    const __bf16* __restrict__ Vb, __bf16* __restrict__ O,
    int S, int NH, int dk) {
  const int b = blockIdx.z, h = blockIdx.y, qt = blockIdx.x;
  const int D = NH * dk;
  const int tid = threadIdx.x, lane = tid & 31, w = tid >> 5;
  const int ll = lane & 15, hi = lane >> 4;
  const float scale = 1.f / sqrtf((float)dk);

  __shared__ __bf16 sK[32][72];        // [key][dk], 144B stride (16B aligned)
  __shared__ __bf16 sV[64][40];        // transposed: [dk][key]
  __shared__ __bf16 sP[4][16][40];     // per-wave P tile [q][key]

  // Q A-fragments held in registers for the whole kernel (2 dk-chunks of 32)
  const int qrow = qt * 64 + w * 16 + ll;
  const __bf16* qp = Q + ((size_t)b * S + qrow) * D + h * dk;
  bf16x16 aq[2];
  aq[0] = frag_ld(qp + hi * 8,       qp + 16 + hi * 8);
  aq[1] = frag_ld(qp + 32 + hi * 8,  qp + 48 + hi * 8);

  f32x8 accO[4];
#pragma unroll
  for (int t = 0; t < 4; ++t) accO[t] = zero8();
  float mrow[8], lrow[8];
#pragma unroll
  for (int i = 0; i < 8; ++i) { mrow[i] = -3.0e38f; lrow[i] = 0.f; }

  const int kr = tid >> 2, seg = (tid & 3) * 16;   // staging assignment

  for (int k0 = 0; k0 < S; k0 += 32) {
    // ---- stage K (row-major) and V (transposed) into LDS ----
    const __bf16* kp = Kb + ((size_t)b * S + k0 + kr) * D + h * dk + seg;
    *(bf16x8*)&sK[kr][seg]     = *(const bf16x8*)kp;
    *(bf16x8*)&sK[kr][seg + 8] = *(const bf16x8*)(kp + 8);
    const __bf16* vp = Vb + ((size_t)b * S + k0 + kr) * D + h * dk + seg;
    bf16x8 v0 = *(const bf16x8*)vp, v1 = *(const bf16x8*)(vp + 8);
#pragma unroll
    for (int j = 0; j < 8; ++j) { sV[seg + j][kr] = v0[j]; sV[seg + 8 + j][kr] = v1[j]; }
    __syncthreads();

    // ---- scores: S(16q x 32k) = Q(16x64) * K^T ----
    f32x8 sc[2];
#pragma unroll
    for (int j = 0; j < 2; ++j) {
      const __bf16* kprow = &sK[j * 16 + ll][0];
      bf16x16 b0 = frag_ld(kprow + hi * 8,      kprow + 16 + hi * 8);
      bf16x16 b1 = frag_ld(kprow + 32 + hi * 8, kprow + 48 + hi * 8);
      f32x8 s = zero8();
      s = WMMA_BF16(aq[0], b0, s);
      s = WMMA_BF16(aq[1], b1, s);
      sc[j] = s;
    }

    // ---- online softmax update (rows live in 16-lane halves) ----
#pragma unroll
    for (int i = 0; i < 8; ++i) {
      float s0 = sc[0][i] * scale, s1 = sc[1][i] * scale;
      float mx = red_max16(fmaxf(s0, s1));
      float mnew = fmaxf(mrow[i], mx);
      float alpha = __expf(mrow[i] - mnew);
      float p0 = __expf(s0 - mnew), p1 = __expf(s1 - mnew);
      float rs = red_sum16(p0 + p1);
      lrow[i] = lrow[i] * alpha + rs;
      mrow[i] = mnew;
#pragma unroll
      for (int t = 0; t < 4; ++t) accO[t][i] = accO[t][i] * alpha;
      const int r = i + 8 * hi;                      // C-layout row
      sP[w][r][ll]      = (__bf16)p0;                // keys 0..15
      sP[w][r][16 + ll] = (__bf16)p1;                // keys 16..31
    }
    // same-wave LDS store->load is in-order; compiler inserts dscnt wait

    // ---- O += P(16x32) * V(32x64) ----
    const __bf16* pp = &sP[w][ll][0];
    bf16x16 pa = frag_ld(pp + hi * 8, pp + 16 + hi * 8);
#pragma unroll
    for (int t = 0; t < 4; ++t) {
      const __bf16* vrow = &sV[t * 16 + ll][0];
      bf16x16 bv = frag_ld(vrow + hi * 8, vrow + 16 + hi * 8);
      accO[t] = WMMA_BF16(pa, bv, accO[t]);
    }
    __syncthreads();   // before next chunk restages sK/sV
  }

  // ---- write out O / l ----
  __bf16* op = O + ((size_t)b * S) * D + h * dk;
#pragma unroll
  for (int t = 0; t < 4; ++t) {
#pragma unroll
    for (int i = 0; i < 8; ++i) {
      const int q = qt * 64 + w * 16 + i + 8 * hi;
      op[(size_t)q * D + t * 16 + ll] = (__bf16)(accO[t][i] / lrow[i]);
    }
  }
}

// ---------------------------------------------------------------------------
// Final: LN(last token) -> feat @ Wfc + bfc -> softmax.  One block per batch.
// ---------------------------------------------------------------------------
__global__ __launch_bounds__(256) void head_ln_fc_softmax(
    const float* __restrict__ Hbuf, const float* __restrict__ gF,
    const float* __restrict__ bF, const float* __restrict__ Wfc,
    const float* __restrict__ bfc, float* __restrict__ out,
    int S, int D, int O) {
  __shared__ float sred[8];
  __shared__ float feat[1024];
  __shared__ float logits[16];
  const int b = blockIdx.x, tid = threadIdx.x;
  const float* x = Hbuf + ((size_t)b * S + (S - 1)) * (size_t)D;

  float ps = 0.f;
  for (int d = tid; d < D; d += 256) ps += x[d];
  const float mean = block_sum256(ps, sred) / (float)D;
  float pv = 0.f;
  for (int d = tid; d < D; d += 256) { float t = x[d] - mean; pv += t * t; }
  const float var = block_sum256(pv, sred) / (float)(D - 1);
  const float rstd = 1.f / (sqrtf(var) + 1e-6f);
  for (int d = tid; d < D; d += 256)
    feat[d] = gF[d] * (x[d] - mean) * rstd + bF[d];
  __syncthreads();

  const int w = tid >> 5, lane = tid & 31;
#pragma unroll
  for (int j = 0; j < 2; ++j) {
    const int o = w * 2 + j;
    float s = 0.f;
    for (int d = lane; d < D; d += 32) s += feat[d] * Wfc[(size_t)d * O + o];
#pragma unroll
    for (int m = 1; m < 32; m <<= 1) s += __shfl_xor(s, m, 32);
    if (lane == 0) logits[o] = s + bfc[o];
  }
  __syncthreads();
  if (tid == 0) {
    float mx = -3.0e38f;
    for (int o = 0; o < O; ++o) mx = fmaxf(mx, logits[o]);
    float ssum = 0.f, e[16];
    for (int o = 0; o < O; ++o) { e[o] = __expf(logits[o] - mx); ssum += e[o]; }
    for (int o = 0; o < O; ++o) out[(size_t)b * O + o] = e[o] / ssum;
  }
}

// ---------------------------------------------------------------------------
// Host orchestration
// ---------------------------------------------------------------------------
extern "C" void kernel_launch(void* const* d_in, const int* in_sizes, int n_in,
                              void* d_out, int out_size, void* d_ws, size_t ws_size,
                              hipStream_t stream) {
  const int B = 8, S = 1024, D = 1024, F = 4096, L = 6, NH = 16, O = 16;
  const size_t M = (size_t)B * S;

  const float* x1   = (const float*)d_in[0];
  const float* Wq   = (const float*)d_in[1];
  const float* bq   = (const float*)d_in[2];
  const float* Wk   = (const float*)d_in[3];
  const float* bk   = (const float*)d_in[4];
  const float* Wv   = (const float*)d_in[5];
  const float* bv   = (const float*)d_in[6];
  const float* Wo   = (const float*)d_in[7];
  const float* bo   = (const float*)d_in[8];
  const float* g0   = (const float*)d_in[9];
  const float* c0   = (const float*)d_in[10];
  const float* W1   = (const float*)d_in[11];
  const float* b1   = (const float*)d_in[12];
  const float* W2   = (const float*)d_in[13];
  const float* b2   = (const float*)d_in[14];
  const float* g1   = (const float*)d_in[15];
  const float* c1   = (const float*)d_in[16];
  const float* gF   = (const float*)d_in[17];
  const float* bF   = (const float*)d_in[18];
  const float* Wfc  = (const float*)d_in[19];
  const float* bfc  = (const float*)d_in[20];

  // Workspace carve-out (256B aligned). Total 320 MiB.
  size_t off = 0;
  auto take = [&](size_t bytes) -> void* {
    void* r = (char*)d_ws + off;
    off += (bytes + 255) & ~(size_t)255;
    return r;
  };
  __bf16* wqT  = (__bf16*)take((size_t)L * D * D * 2);
  __bf16* wkT  = (__bf16*)take((size_t)L * D * D * 2);
  __bf16* wvT  = (__bf16*)take((size_t)L * D * D * 2);
  __bf16* woT  = (__bf16*)take((size_t)L * D * D * 2);
  __bf16* w1T  = (__bf16*)take((size_t)L * D * F * 2);   // [F][D] per layer
  __bf16* w2T  = (__bf16*)take((size_t)L * D * F * 2);   // [D][F] per layer
  float*  hbuf = (float*) take(M * D * 4);
  __bf16* ybuf = (__bf16*)take(M * D * 2);
  __bf16* qb   = (__bf16*)take(M * D * 2);
  __bf16* kbuf = (__bf16*)take(M * D * 2);
  __bf16* vbuf = (__bf16*)take(M * D * 2);
  __bf16* ab   = (__bf16*)take(M * D * 2);
  __bf16* midb = (__bf16*)take(M * F * 2);
  (void)ws_size; (void)n_in; (void)in_sizes; (void)out_size;

  // h = x
  hipMemcpyAsync(hbuf, x1, M * D * sizeof(float), hipMemcpyDeviceToDevice, stream);

  // Weight convert+transpose (runs every call; tiny vs GEMM cost, deterministic)
  for (int l = 0; l < L; ++l) {
    const size_t dd = (size_t)l * D * D, df = (size_t)l * D * F;
    wconv_tr<<<dim3(D / 32, D / 32), 256, 0, stream>>>(Wq + dd, wqT + dd, D, D);
    wconv_tr<<<dim3(D / 32, D / 32), 256, 0, stream>>>(Wk + dd, wkT + dd, D, D);
    wconv_tr<<<dim3(D / 32, D / 32), 256, 0, stream>>>(Wv + dd, wvT + dd, D, D);
    wconv_tr<<<dim3(D / 32, D / 32), 256, 0, stream>>>(Wo + dd, woT + dd, D, D);
    wconv_tr<<<dim3(F / 32, D / 32), 256, 0, stream>>>(W1 + df, w1T + df, D, F); // [D][F]->[F][D]
    wconv_tr<<<dim3(D / 32, F / 32), 256, 0, stream>>>(W2 + df, w2T + df, F, D); // [F][D]->[D][F]
  }

  const dim3 gD(D / BN, M / BM);   // N=1024 GEMMs
  const dim3 gF4(F / BN, M / BM);  // N=4096 GEMM
  for (int l = 0; l < L; ++l) {
    const size_t dd = (size_t)l * D * D, df = (size_t)l * D * F;
    // y = LN0(h)
    ln_to_bf16<<<(int)M, 256, 0, stream>>>(hbuf, g0 + (size_t)l * D, c0 + (size_t)l * D, ybuf, D);
    // q,k,v = y @ W{q,k,v} + b
    gemm_bf16_wmma<<<gD, 256, 0, stream>>>(ybuf, wqT + dd, bq + (size_t)l * D, qb,   nullptr, (int)M, D, D, 0);
    gemm_bf16_wmma<<<gD, 256, 0, stream>>>(ybuf, wkT + dd, bk + (size_t)l * D, kbuf, nullptr, (int)M, D, D, 0);
    gemm_bf16_wmma<<<gD, 256, 0, stream>>>(ybuf, wvT + dd, bv + (size_t)l * D, vbuf, nullptr, (int)M, D, D, 0);
    // a = softmax(qk^T/sqrt(dk)) v
    attn_wmma<<<dim3(S / 64, NH, B), 128, 0, stream>>>(qb, kbuf, vbuf, ab, S, NH, D / NH);
    // h = h + a @ Wo + bo
    gemm_bf16_wmma<<<gD, 256, 0, stream>>>(ab, woT + dd, bo + (size_t)l * D, hbuf, hbuf, (int)M, D, D, 2);
    // y = LN1(h)
    ln_to_bf16<<<(int)M, 256, 0, stream>>>(hbuf, g1 + (size_t)l * D, c1 + (size_t)l * D, ybuf, D);
    // mid = relu(y @ W1 + b1) ;  h = h + mid @ W2 + b2
    gemm_bf16_wmma<<<gF4, 256, 0, stream>>>(ybuf, w1T + df, b1 + (size_t)l * F, midb, nullptr, (int)M, F, D, 1);
    gemm_bf16_wmma<<<gD,  256, 0, stream>>>(midb, w2T + df, b2 + (size_t)l * D, hbuf, hbuf, (int)M, D, F, 2);
  }

  head_ln_fc_softmax<<<B, 256, 0, stream>>>(hbuf, gF, bF, Wfc, bfc, (float*)d_out, S, D, O);
}
